// DubinsLSTM_38397007627002
// MI455X (gfx1250) — matrix operand order repaired and checked by reading, not compiled, not run on hardware
//
#include <hip/hip_runtime.h>
#include <stdint.h>

// ---------------------------------------------------------------------------
// DubinsLSTM on gfx1250: persistent batch-tiled LSTM with v_wmma_f32_16x16x32_f16
//   grid = 64 workgroups (B/16), block = 256 threads (8 waves, wave32)
//   wave w computes gate columns [w*64, w*64+64) of the 512-wide z for its
//   16-row batch tile; weights are pre-packed into WMMA-B fragment order and
//   streamed from L2 every step via global_load_b128 (pointer laundered so the
//   compiler cannot hoist them across the t-loop and spill; explicitly cast to
//   addrspace(1) so they stay on the GLOBAL path, not FLAT).
// ---------------------------------------------------------------------------

typedef __attribute__((ext_vector_type(16))) _Float16 v16h;
typedef __attribute__((ext_vector_type(8)))  float    v8f;
typedef __attribute__((ext_vector_type(4)))  unsigned v4u;
typedef __attribute__((address_space(1))) const v4u gv4u;   // global-memory fragment ptr

union FragV { v4u q[2]; v16h h; };   // 8 VGPRs = one 16x32 (A) / 32x16 (B) f16 fragment

#define BB   1024
#define TT   200
#define HH   128
#define N4H  512      // 4*H
#define NT   32       // N tiles of 16 in 512
#define KT0  9        // ceil(259/32): K = prev(3)+cond(128)+h0(128), padded to 288
#define KT1  8        // 256/32: K = h0(128)+h1(128)
#define KP0  288
#define KP1  256
#define ZROW 520      // f32 z-buffer row stride (520 % 64 = 8 -> no bank pileup)

// ---------------- weight pre-pack: f32 [K][512] -> f16 B-fragment order ----
// packed dword index: ((kt*NT + nt)*32 + lane)*8 + j
//   K = kt*32 + (lane/16)*16 + 2j (+0/+1 in lo/hi f16), N = nt*16 + lane%16
__global__ __launch_bounds__(256) void pack_weights(const float* __restrict__ Wih,
                                                    const float* __restrict__ Whh,
                                                    int kih, int KT,
                                                    unsigned* __restrict__ packed) {
    int tid = blockIdx.x * 256 + threadIdx.x;
    int total = KT * NT * 32 * 8;
    if (tid >= total) return;
    int j    = tid & 7;
    int L    = (tid >> 3) & 31;
    int frag = tid >> 8;
    int nt   = frag & (NT - 1);
    int kt   = frag >> 5;
    int K    = kt * 32 + (L >> 4) * 16 + 2 * j;
    int N    = nt * 16 + (L & 15);
    auto fetch = [&](int k) -> float {
        if (k < kih)       return Wih[(size_t)k * N4H + N];
        if (k < kih + HH)  return Whh[(size_t)(k - kih) * N4H + N];
        return 0.0f;
    };
    _Float16 lo = (_Float16)fetch(K);
    _Float16 hi = (_Float16)fetch(K + 1);
    unsigned u = ((unsigned)__builtin_bit_cast(unsigned short, hi) << 16) |
                 (unsigned)__builtin_bit_cast(unsigned short, lo);
    packed[tid] = u;
}

__device__ __forceinline__ float sigmoidf_(float x) {
    return 1.0f / (1.0f + __expf(-x));
}
__device__ __forceinline__ float tanhf_(float x) {
    // overflow-safe: tanh(x) = sign(x) * (1 - e) / (1 + e), e = exp(-2|x|)
    float e = __expf(-2.0f * fabsf(x));
    float r = (1.0f - e) / (1.0f + e);
    return copysignf(r, x);
}

// ------------------------------- main kernel -------------------------------
__global__ __launch_bounds__(256) void lstm_persistent(
    const float* __restrict__ conds,    // [B,4]
    const float* __restrict__ tgt,      // [B,T,3]
    const int*   __restrict__ lengths,  // [B]
    const float* __restrict__ tfr,      // [T,B]
    const float* __restrict__ Wc,       // [4,128]
    const float* __restrict__ bc,       // [128]
    const float* __restrict__ b0,       // [512]
    const float* __restrict__ b1,       // [512]
    const float* __restrict__ Wo,       // [128,3]
    const float* __restrict__ bo,       // [3]
    const unsigned* pk0,                // packed layer-0 weights (K=288)
    const unsigned* pk1,                // packed layer-1 weights (K=256)
    float* __restrict__ out)            // [B,T,3]
{
    __shared__ alignas(16) _Float16 Xh[16][KP0];   // prev(0..2) | cond(3..130) | h0(131..258) | pad
    __shared__ alignas(16) _Float16 X1[16][KP1];   // h0(0..127) | h1(128..255)
    __shared__ float zb[16][ZROW];
    __shared__ float c0s[16][HH], c1s[16][HH], h1s[16][HH];
    __shared__ float b0s[N4H], b1s[N4H], Wos[HH * 3], bos[3];
    __shared__ float predp[16][3][4];

    const int tid  = threadIdx.x;
    const int lane = tid & 31;
    const int wave = tid >> 5;
    const int m0   = blockIdx.x * 16;
    const int mrow = lane & 15;
    const int half = lane >> 4;

    // ------------- one-time init -------------
    for (int i = tid; i < N4H; i += 256) { b0s[i] = b0[i]; b1s[i] = b1[i]; }
    for (int i = tid; i < HH * 3; i += 256) Wos[i] = Wo[i];
    if (tid < 3) bos[tid] = bo[tid];
    for (int i = tid; i < 16 * KP0; i += 256) ((_Float16*)Xh)[i] = (_Float16)0.0f;
    for (int i = tid; i < 16 * KP1; i += 256) ((_Float16*)X1)[i] = (_Float16)0.0f;
    for (int i = tid; i < 16 * HH; i += 256) {
        int r = i >> 7, c = i & (HH - 1);
        c0s[r][c] = 0.0f; c1s[r][c] = 0.0f;
    }
    __syncthreads();
    // cond_embed = conds @ Wc + bc  (tiny: 4-term dot), written once as f16
    for (int i = tid; i < 16 * HH; i += 256) {
        int r = i >> 7, c = i & (HH - 1);
        float acc = bc[c];
#pragma unroll
        for (int k = 0; k < 4; ++k) acc += conds[(size_t)(m0 + r) * 4 + k] * Wc[k * HH + c];
        Xh[r][3 + c] = (_Float16)acc;
    }
    __syncthreads();

    const unsigned* p0 = pk0;
    const unsigned* p1 = pk1;

    for (int t = 0; t < TT; ++t) {
        // Launder the weight base pointers: zero-cost SGPR no-op that forbids
        // hoisting the (loop-invariant) B-fragment loads across the t-loop
        // (which previously caused VGPR blowup + scratch spills). Then cast
        // into addrspace(1) so the loads lower to global_load_b128 (LOADcnt
        // only) instead of flat_load_b128 (LOADcnt + DScnt, ties up LDS pipe).
        asm volatile("" : "+s"(p0), "+s"(p1));
        gv4u* p0v = (gv4u*)(uintptr_t)p0;
        gv4u* p1v = (gv4u*)(uintptr_t)p1;

        // ================= layer-0 GEMM: z = [16,288] @ [288,512] ==========
        {
            v8f acc[4] = {v8f{}, v8f{}, v8f{}, v8f{}};
            for (int kt = 0; kt < KT0; ++kt) {
                FragV a;
                a.q[0] = *reinterpret_cast<const v4u*>(&Xh[mrow][kt * 32 + half * 16]);
                a.q[1] = *reinterpret_cast<const v4u*>(&Xh[mrow][kt * 32 + half * 16 + 8]);
#pragma unroll
                for (int n = 0; n < 4; ++n) {
                    size_t idx = ((size_t)(kt * NT + wave * 4 + n) * 32 + lane) * 2;
                    FragV b; b.q[0] = p0v[idx]; b.q[1] = p0v[idx + 1];
                    acc[n] = __builtin_amdgcn_wmma_f32_16x16x32_f16(
                        false, a.h, false, b.h, (short)0, acc[n], false, false);
                }
            }
#pragma unroll
            for (int n = 0; n < 4; ++n) {
                int ncol = wave * 64 + n * 16 + mrow;
#pragma unroll
                for (int r = 0; r < 8; ++r) zb[r + 8 * half][ncol] = acc[n][r];
            }
        }
        __syncthreads();
        // ---------------- layer-0 gates / state update ---------------------
        for (int i = tid; i < 16 * HH; i += 256) {
            int r = i >> 7, j = i & (HH - 1);
            float zi = zb[r][j]            + b0s[j];
            float zf = zb[r][HH + j]       + b0s[HH + j];
            float zg = zb[r][2 * HH + j]   + b0s[2 * HH + j];
            float zo = zb[r][3 * HH + j]   + b0s[3 * HH + j];
            float cN = sigmoidf_(zf) * c0s[r][j] + sigmoidf_(zi) * tanhf_(zg);
            float hN = sigmoidf_(zo) * tanhf_(cN);
            c0s[r][j] = cN;
            X1[r][j]        = (_Float16)hN;   // input to layer 1 (this step)
            Xh[r][131 + j]  = (_Float16)hN;   // recurrent input (next step)
        }
        __syncthreads();
        // ================= layer-1 GEMM: z = [16,256] @ [256,512] ==========
        {
            v8f acc[4] = {v8f{}, v8f{}, v8f{}, v8f{}};
            for (int kt = 0; kt < KT1; ++kt) {
                FragV a;
                a.q[0] = *reinterpret_cast<const v4u*>(&X1[mrow][kt * 32 + half * 16]);
                a.q[1] = *reinterpret_cast<const v4u*>(&X1[mrow][kt * 32 + half * 16 + 8]);
#pragma unroll
                for (int n = 0; n < 4; ++n) {
                    size_t idx = ((size_t)(kt * NT + wave * 4 + n) * 32 + lane) * 2;
                    FragV b; b.q[0] = p1v[idx]; b.q[1] = p1v[idx + 1];
                    acc[n] = __builtin_amdgcn_wmma_f32_16x16x32_f16(
                        false, a.h, false, b.h, (short)0, acc[n], false, false);
                }
            }
#pragma unroll
            for (int n = 0; n < 4; ++n) {
                int ncol = wave * 64 + n * 16 + mrow;
#pragma unroll
                for (int r = 0; r < 8; ++r) zb[r + 8 * half][ncol] = acc[n][r];
            }
        }
        __syncthreads();
        // ---------------- layer-1 gates / state update ---------------------
        for (int i = tid; i < 16 * HH; i += 256) {
            int r = i >> 7, j = i & (HH - 1);
            float zi = zb[r][j]            + b1s[j];
            float zf = zb[r][HH + j]       + b1s[HH + j];
            float zg = zb[r][2 * HH + j]   + b1s[2 * HH + j];
            float zo = zb[r][3 * HH + j]   + b1s[3 * HH + j];
            float cN = sigmoidf_(zf) * c1s[r][j] + sigmoidf_(zi) * tanhf_(zg);
            float hN = sigmoidf_(zo) * tanhf_(cN);
            c1s[r][j] = cN;
            h1s[r][j] = hN;                   // f32 copy for the output head
            X1[r][HH + j] = (_Float16)hN;     // recurrent input (next step)
        }
        __syncthreads();
        // ---------------- output head (4-way split dot) --------------------
        if (tid < 192) {                       // 16 rows x 3 dims x 4 quarters
            int r = tid / 12;
            int d = (tid / 4) % 3;
            int q = tid & 3;
            float s = 0.0f;
#pragma unroll
            for (int j = q * 32; j < q * 32 + 32; ++j) s += h1s[r][j] * Wos[j * 3 + d];
            predp[r][d][q] = s;
        }
        __syncthreads();
        // ---------------- reduce + write + teacher forcing -----------------
        if (tid < 48) {
            int r = tid / 3, d = tid - 3 * (tid / 3);
            float p = bos[d] + predp[r][d][0] + predp[r][d][1] + predp[r][d][2] + predp[r][d][3];
            int b = m0 + r;
            out[(size_t)b * TT * 3 + t * 3 + d] = p;
            bool use_tf = (tfr[(size_t)t * BB + b] < 0.5f) && (t < lengths[b]);
            float pv = use_tf ? tgt[(size_t)b * TT * 3 + t * 3 + d] : p;
            Xh[r][d] = (_Float16)pv;          // prev for next step
        }
        __syncthreads();
    }
}

// ---------------------------------------------------------------------------
extern "C" void kernel_launch(void* const* d_in, const int* in_sizes, int n_in,
                              void* d_out, int out_size, void* d_ws, size_t ws_size,
                              hipStream_t stream) {
    (void)in_sizes; (void)n_in; (void)out_size; (void)ws_size;
    const float* conds   = (const float*)d_in[0];
    const float* tgt     = (const float*)d_in[1];
    const int*   lengths = (const int*)  d_in[2];
    const float* tfr     = (const float*)d_in[3];
    const float* Wc      = (const float*)d_in[4];
    const float* bc      = (const float*)d_in[5];
    const float* W_ih0   = (const float*)d_in[6];
    const float* W_hh0   = (const float*)d_in[7];
    const float* b0      = (const float*)d_in[8];
    const float* W_ih1   = (const float*)d_in[9];
    const float* W_hh1   = (const float*)d_in[10];
    const float* b1      = (const float*)d_in[11];
    const float* Wo      = (const float*)d_in[12];
    const float* bo      = (const float*)d_in[13];
    float* out = (float*)d_out;

    // workspace: packed f16 weights (fragment order). layer0: 288x512, layer1: 256x512
    unsigned* pk0 = (unsigned*)d_ws;                                  // 73728 dwords
    unsigned* pk1 = (unsigned*)((char*)d_ws + (size_t)KT0 * NT * 32 * 8 * 4);

    pack_weights<<<KT0 * NT, 256, 0, stream>>>(W_ih0, W_hh0, 131, KT0, pk0);
    pack_weights<<<KT1 * NT, 256, 0, stream>>>(W_ih1, W_hh1, 128, KT1, pk1);

    lstm_persistent<<<BB / 16, 256, 0, stream>>>(
        conds, tgt, lengths, tfr, Wc, bc, b0, b1, Wo, bo, pk0, pk1, out);
}
